// SCAM_3109556322745
// MI455X (gfx1250) — matrix operand order repaired and checked by reading, not compiled, not run on hardware
//
#include <hip/hip_runtime.h>
#include <hip/hip_bf16.h>

// CDNA5 / gfx1250 fused SCAM (cross-attention) kernel.
// One workgroup (8 wave32s) per (b,h) slice; everything fused in 283KB LDS.
// Wave roles selected by scalar (readfirstlane-uniform) branches so EXEC is
// all-ones at every v_wmma (ISA 7.12 requirement).

typedef _Float16 h16;
typedef __attribute__((ext_vector_type(16))) _Float16 v16h;
typedef __attribute__((ext_vector_type(8)))  _Float16 v8h;
typedef __attribute__((ext_vector_type(8)))  float    v8f;
typedef __attribute__((ext_vector_type(4)))  float    v4f;

namespace {
constexpr int Bn = 16, Cn = 64, Hn = 128, Wn = 128;
constexpr int HWc = Hn * Wn;

// LDS row strides (padded; f16 ones keep 16B alignment for b128 fragment loads)
constexpr int LDF = 66;   // f32 x-slices   [128][66]
constexpr int LDH = 72;   // f16 matrices   [.][72]
constexpr int LDSS = 132; // f32 scores     [128][132]
constexpr int LDT = 136;  // f16 V^T        [64][136]

constexpr size_t SMEM_F32 = (size_t)(2 * 128 * LDF + 128 * LDSS + 4 * 128) * 4;
constexpr size_t SMEM_H16 = (size_t)(4 * 128 * LDH + 4 * 64 * LDH + 2 * 64 * LDT) * 2;
constexpr size_t SMEM_BYTES = SMEM_F32 + SMEM_H16; // = 282,624 bytes (< 320KB WGP LDS)
} // namespace

__device__ __forceinline__ v8f wmma_f16(v16h a, v16h b, v8f c) {
  return __builtin_amdgcn_wmma_f32_16x16x32_f16(
      /*neg_a=*/false, a, /*neg_b=*/false, b,
      /*c_mod=*/(short)0, c, /*reuse_a=*/false, /*reuse_b=*/false);
}

// ---- A-operand fragment (16x32 f16): per-lane split-K pattern (ISA 7.12.2) ----
__device__ __forceinline__ v16h frag_a_h(const h16* base, int r0, int k0, int ld) {
  const int lane = threadIdx.x & 31;
  const int r = r0 + (lane & 15);
  const int kb = k0 + ((lane >> 4) << 3);      // +8 halves for upper lane half
  const h16* p = base + r * ld + kb;
  v8h lo = *(const v8h*)(p);
  v8h hi = *(const v8h*)(p + 16);
  v16h f;
#pragma unroll
  for (int i = 0; i < 8; ++i) { f[i] = lo[i]; f[i + 8] = hi[i]; }
  return f;
}

// A fragment sourced from f32 LDS (converted on the fly)
__device__ __forceinline__ v16h frag_a_f32(const float* base, int r0, int k0, int ld) {
  const int lane = threadIdx.x & 31;
  const int r = r0 + (lane & 15);
  const int kb = k0 + ((lane >> 4) << 3);
  const float* p = base + r * ld + kb;
  v16h f;
#pragma unroll
  for (int i = 0; i < 8; ++i) { f[i] = (h16)p[i]; f[i + 8] = (h16)p[i + 16]; }
  return f;
}

// A fragment = row-softmax of S, generated on the fly (attn_h)
__device__ __forceinline__ v16h frag_a_smax_row(const float* S, int r0, int k0,
                                                const float* rmax, const float* rinv) {
  const int lane = threadIdx.x & 31;
  const int r = r0 + (lane & 15);
  const int kb = k0 + ((lane >> 4) << 3);
  const float m = rmax[r], inv = rinv[r];
  const float* p = S + r * LDSS + kb;
  v16h f;
#pragma unroll
  for (int i = 0; i < 8; ++i) {
    f[i]     = (h16)(__expf(p[i]      - m) * inv);
    f[i + 8] = (h16)(__expf(p[i + 16] - m) * inv);
  }
  return f;
}

// A fragment = column-softmax of S read transposed (attn_l)
__device__ __forceinline__ v16h frag_a_smax_col(const float* S, int r0, int k0,
                                                const float* cmax, const float* cinv) {
  const int lane = threadIdx.x & 31;
  const int r = r0 + (lane & 15);              // output row w == column of S
  const int kb = k0 + ((lane >> 4) << 3);
  const float m = cmax[r], inv = cinv[r];
  v16h f;
#pragma unroll
  for (int i = 0; i < 8; ++i) {
    f[i]     = (h16)(__expf(S[(kb + i) * LDSS + r]      - m) * inv);
    f[i + 8] = (h16)(__expf(S[(kb + 16 + i) * LDSS + r] - m) * inv);
  }
  return f;
}

// ---- B-operand fragment (32x16 f16): per-lane split-K pattern (matches A) ----
__device__ __forceinline__ v16h frag_b_h(const h16* base, int n0, int k0, int ld) {
  const int lane = threadIdx.x & 31;
  const int n = n0 + (lane & 15);
  const int kb = k0 + ((lane >> 4) << 3);      // +8 halves for upper lane half
  const h16* p = base + n * ld + kb;
  v8h lo = *(const v8h*)(p);
  v8h hi = *(const v8h*)(p + 16);
  v16h f;
#pragma unroll
  for (int i = 0; i < 8; ++i) { f[i] = lo[i]; f[i + 8] = hi[i]; }
  return f;
}

// ---- D-tile stores (C/D layout: row = m0 + r + 8*(lane>=16), col = lane&15) ----
__device__ __forceinline__ void st_d_f16(h16* dst, int m0, int n0, int ld, v8f d) {
  const int lane = threadIdx.x & 31;
  const int n = n0 + (lane & 15);
  const int rb = m0 + ((lane >> 4) << 3);
#pragma unroll
  for (int r = 0; r < 8; ++r) dst[(rb + r) * ld + n] = (h16)d[r];
}

__device__ __forceinline__ void st_d_f16T(h16* dst, int m0, int n0, int ldT, v8f d) {
  const int lane = threadIdx.x & 31;
  const int n = n0 + (lane & 15);
  const int rb = m0 + ((lane >> 4) << 3);
#pragma unroll
  for (int r = 0; r < 8; ++r) dst[n * ldT + rb + r] = (h16)d[r];
}

__device__ __forceinline__ void st_d_f32_scaled(float* dst, int m0, int n0, int ld,
                                                v8f d, float s) {
  const int lane = threadIdx.x & 31;
  const int n = n0 + (lane & 15);
  const int rb = m0 + ((lane >> 4) << 3);
#pragma unroll
  for (int r = 0; r < 8; ++r) dst[(rb + r) * ld + n] = d[r] * s;
}

__device__ __forceinline__ void acc_d_f32(float* dst, int m0, int n0, int ld, v8f d) {
  const int lane = threadIdx.x & 31;
  const int n = n0 + (lane & 15);
  const int rb = m0 + ((lane >> 4) << 3);
#pragma unroll
  for (int r = 0; r < 8; ++r) dst[(rb + r) * ld + n] += d[r];
}

__global__ __launch_bounds__(256)
void scam_cdna5_kernel(const float* __restrict__ xl_g, const float* __restrict__ xh_g,
                       const float* __restrict__ g1, const float* __restrict__ b1,
                       const float* __restrict__ g2, const float* __restrict__ b2,
                       const float* __restrict__ Wq, const float* __restrict__ Wk,
                       const float* __restrict__ Wv1, const float* __restrict__ Wv2,
                       float* __restrict__ out_l, float* __restrict__ out_h) {
  extern __shared__ char smem_raw[];
  float* xl32 = (float*)smem_raw;              // [128][66]   x_l slice (later xl_out)
  float* xh32 = xl32 + 128 * LDF;              // [128][66]   x_h slice (later xh_out)
  float* S32  = xh32 + 128 * LDF;              // [128][132]  scores (pre-softmax)
  float* rmax = S32 + 128 * LDSS;
  float* rinv = rmax + 128;
  float* cmax = rinv + 128;
  float* cinv = cmax + 128;
  h16* lnl = (h16*)(cinv + 128);               // [128][72] LN(x_l) f16
  h16* lnh = lnl + 128 * LDH;                  // [128][72] LN(x_h) f16
  h16* q16 = lnh + 128 * LDH;                  // [128][72]
  h16* k16 = q16 + 128 * LDH;                  // [128][72]
  h16* Wh  = k16 + 128 * LDH;                  // 4 x [64][72]  (Wq,Wk,Wv1,Wv2 f16)
  h16* vlT = Wh + 4 * 64 * LDH;                // [64][136]  (x_l @ Wv1^T)^T
  h16* vhT = vlT + 64 * LDT;                   // [64][136]  (x_h @ Wv2^T)^T

  const int tid = threadIdx.x;
  // Scalar (SGPR) wave id: guarantees uniform branches and full EXEC at WMMAs.
  const int wvu = __builtin_amdgcn_readfirstlane(tid >> 5);   // 0..7
  const int b   = blockIdx.x / Hn;
  const int h   = blockIdx.x % Hn;
  const long base = (long)b * Cn * HWc + (long)h * Wn;

  // ---------- Phase A: coalesced load of both x slices (float4 along w) ----------
#pragma unroll
  for (int i = 0; i < 8; ++i) {
    const int idx = i * 256 + tid;             // 2048 float4s per tensor
    const int f = idx * 4;
    const int c = f >> 7;
    const int w = f & 127;
    const long g = base + (long)c * HWc + w;
    v4f a = *(const v4f*)(xl_g + g);
    v4f d = *(const v4f*)(xh_g + g);
#pragma unroll
    for (int j = 0; j < 4; ++j) {
      xl32[(w + j) * LDF + c] = a[j];
      xh32[(w + j) * LDF + c] = d[j];
    }
  }
  // Weights -> f16 LDS, layout [c_out][c_in] (== WMMA B layout [n][k])
#pragma unroll
  for (int s = 0; s < 4; ++s) {
    const float* src = (s == 0) ? Wq : (s == 1) ? Wk : (s == 2) ? Wv1 : Wv2;
    h16* dst = Wh + s * 64 * LDH;
#pragma unroll
    for (int i = 0; i < 16; ++i) {
      const int e = i * 256 + tid;             // 0..4095
      dst[(e >> 6) * LDH + (e & 63)] = (h16)src[e];
    }
  }
  __syncthreads();

  // ---------- Phase B: LayerNorm (waves 0-3: x_l rows, waves 4-7: x_h rows) ----------
  {
    const int sel = __builtin_amdgcn_readfirstlane(tid >> 7); // uniform per wave
    const int w = tid & 127;
    const float* src = sel ? xh32 : xl32;
    const float* gam = sel ? g2 : g1;
    const float* bet = sel ? b2 : b1;
    h16* dst = sel ? lnh : lnl;
    float mean = 0.f;
    for (int c = 0; c < Cn; ++c) mean += src[w * LDF + c];
    mean *= (1.0f / Cn);
    float var = 0.f;
    for (int c = 0; c < Cn; ++c) {
      const float d = src[w * LDF + c] - mean;
      var += d * d;
    }
    var *= (1.0f / Cn);
    const float rstd = rsqrtf(var + 1e-5f);
    for (int c = 0; c < Cn; ++c)
      dst[w * LDH + c] = (h16)((src[w * LDF + c] - mean) * rstd * gam[c] + bet[c]);
  }
  __syncthreads();

  // ---------- Stage 1: q = LN(xl)Wq^T, k = LN(xh)Wk^T, vl = xl Wv1^T, vh = xh Wv2^T
  // Wave role = wvu>>1 (uniform scalar branch); 2 waves per output, 16 tiles each.
  {
    const int selw  = wvu >> 1;                // 0:q 1:k 2:vl 3:vh
    const int halfw = wvu & 1;
    if (selw == 0) {
      for (int t = 0; t < 16; ++t) {
        const int e = halfw * 16 + t;
        const int m0 = (e >> 2) << 4, n0 = (e & 3) << 4;
        v8f acc = {};
        acc = wmma_f16(frag_a_h(lnl, m0, 0,  LDH), frag_b_h(Wh, n0, 0,  LDH), acc);
        acc = wmma_f16(frag_a_h(lnl, m0, 32, LDH), frag_b_h(Wh, n0, 32, LDH), acc);
        st_d_f16(q16, m0, n0, LDH, acc);
      }
    } else if (selw == 1) {
      const h16* Wkh = Wh + 1 * 64 * LDH;
      for (int t = 0; t < 16; ++t) {
        const int e = halfw * 16 + t;
        const int m0 = (e >> 2) << 4, n0 = (e & 3) << 4;
        v8f acc = {};
        acc = wmma_f16(frag_a_h(lnh, m0, 0,  LDH), frag_b_h(Wkh, n0, 0,  LDH), acc);
        acc = wmma_f16(frag_a_h(lnh, m0, 32, LDH), frag_b_h(Wkh, n0, 32, LDH), acc);
        st_d_f16(k16, m0, n0, LDH, acc);
      }
    } else if (selw == 2) {
      const h16* Wv1h = Wh + 2 * 64 * LDH;
      for (int t = 0; t < 16; ++t) {
        const int e = halfw * 16 + t;
        const int m0 = (e >> 2) << 4, n0 = (e & 3) << 4;
        v8f acc = {};
        acc = wmma_f16(frag_a_f32(xl32, m0, 0,  LDF), frag_b_h(Wv1h, n0, 0,  LDH), acc);
        acc = wmma_f16(frag_a_f32(xl32, m0, 32, LDF), frag_b_h(Wv1h, n0, 32, LDH), acc);
        st_d_f16T(vlT, m0, n0, LDT, acc);      // store transposed: [c][v]
      }
    } else {
      const h16* Wv2h = Wh + 3 * 64 * LDH;
      for (int t = 0; t < 16; ++t) {
        const int e = halfw * 16 + t;
        const int m0 = (e >> 2) << 4, n0 = (e & 3) << 4;
        v8f acc = {};
        acc = wmma_f16(frag_a_f32(xh32, m0, 0,  LDF), frag_b_h(Wv2h, n0, 0,  LDH), acc);
        acc = wmma_f16(frag_a_f32(xh32, m0, 32, LDF), frag_b_h(Wv2h, n0, 32, LDH), acc);
        st_d_f16T(vhT, m0, n0, LDT, acc);
      }
    }
  }
  __syncthreads();

  // ---------- Stage 2: S = (q k^T) / sqrt(C); wave wvu owns column strip n0=wvu*16
  {
    const int n0 = wvu << 4;
    for (int t = 0; t < 8; ++t) {
      const int m0 = t << 4;
      v8f acc = {};
      acc = wmma_f16(frag_a_h(q16, m0, 0,  LDH), frag_b_h(k16, n0, 0,  LDH), acc);
      acc = wmma_f16(frag_a_h(q16, m0, 32, LDH), frag_b_h(k16, n0, 32, LDH), acc);
      st_d_f32_scaled(S32, m0, n0, LDSS, acc, 0.125f);     // 1/sqrt(64)
    }
  }
  __syncthreads();

  // ---------- Softmax stats: waves 0-3 rows, waves 4-7 columns ----------
  {
    const int sel = __builtin_amdgcn_readfirstlane(tid >> 7);
    const int x = tid & 127;
    if (sel == 0) {
      float m = -3.4e38f;
      for (int v = 0; v < 128; ++v) m = fmaxf(m, S32[x * LDSS + v]);
      float s = 0.f;
      for (int v = 0; v < 128; ++v) s += __expf(S32[x * LDSS + v] - m);
      rmax[x] = m; rinv[x] = 1.0f / s;
    } else {
      float m = -3.4e38f;
      for (int v = 0; v < 128; ++v) m = fmaxf(m, S32[v * LDSS + x]);
      float s = 0.f;
      for (int v = 0; v < 128; ++v) s += __expf(S32[v * LDSS + x] - m);
      cmax[x] = m; cinv[x] = 1.0f / s;
    }
  }
  __syncthreads();

  // ---------- Stage 3: xl_out = xl + softmax_row(S) @ vh ; xh_out = xh + softmax_col @ vl
  // Wave role = wvu>>2 (uniform); 4 waves per output, 8 tiles each; K=128 => 4 WMMA.
  {
    const int which = wvu >> 2;                // 0: attn_h@vh -> xl ; 1: attn_l@vl -> xh
    const int sub   = wvu & 3;
    if (which == 0) {
      for (int t = 0; t < 8; ++t) {
        const int e = sub * 8 + t;             // 0..31
        const int m0 = (e >> 2) << 4, n0 = (e & 3) << 4;
        v8f acc = {};
#pragma unroll
        for (int kk = 0; kk < 4; ++kk) {
          const int k0 = kk * 32;
          acc = wmma_f16(frag_a_smax_row(S32, m0, k0, rmax, rinv),
                         frag_b_h(vhT, n0, k0, LDT), acc);
        }
        acc_d_f32(xl32, m0, n0, LDF, acc);     // residual in place
      }
    } else {
      for (int t = 0; t < 8; ++t) {
        const int e = sub * 8 + t;
        const int m0 = (e >> 2) << 4, n0 = (e & 3) << 4;
        v8f acc = {};
#pragma unroll
        for (int kk = 0; kk < 4; ++kk) {
          const int k0 = kk * 32;
          acc = wmma_f16(frag_a_smax_col(S32, m0, k0, cmax, cinv),
                         frag_b_h(vlT, n0, k0, LDT), acc);
        }
        acc_d_f32(xh32, m0, n0, LDF, acc);
      }
    }
  }
  __syncthreads();

  // ---------- Phase D: coalesced float4 stores of both outputs ----------
#pragma unroll
  for (int i = 0; i < 8; ++i) {
    const int idx = i * 256 + tid;
    const int f = idx * 4;
    const int c = f >> 7;
    const int w = f & 127;
    const long g = base + (long)c * HWc + w;
    v4f a, d;
#pragma unroll
    for (int j = 0; j < 4; ++j) {
      a[j] = xl32[(w + j) * LDF + c];
      d[j] = xh32[(w + j) * LDF + c];
    }
    *(v4f*)(out_l + g) = a;
    *(v4f*)(out_h + g) = d;
  }
}

extern "C" void kernel_launch(void* const* d_in, const int* in_sizes, int n_in,
                              void* d_out, int out_size, void* d_ws, size_t ws_size,
                              hipStream_t stream) {
  (void)in_sizes; (void)n_in; (void)out_size; (void)d_ws; (void)ws_size;
  const float* xl  = (const float*)d_in[0];
  const float* xh  = (const float*)d_in[1];
  const float* g1  = (const float*)d_in[2];
  const float* b1  = (const float*)d_in[3];
  const float* g2  = (const float*)d_in[4];
  const float* b2  = (const float*)d_in[5];
  const float* Wq  = (const float*)d_in[6];
  const float* Wk  = (const float*)d_in[7];
  const float* Wv1 = (const float*)d_in[8];
  const float* Wv2 = (const float*)d_in[9];
  float* out_l = (float*)d_out;
  float* out_h = out_l + (size_t)Bn * Cn * HWc;

  // Opt in to >64KB dynamic LDS (CDNA5 WGP has 320KB). Deterministic per call;
  // not a stream operation, so graph capture is unaffected.
  (void)hipFuncSetAttribute((const void*)scam_cdna5_kernel,
                            hipFuncAttributeMaxDynamicSharedMemorySize,
                            (int)SMEM_BYTES);

  dim3 grid(Bn * Hn);
  dim3 block(256);
  scam_cdna5_kernel<<<grid, block, SMEM_BYTES, stream>>>(
      xl, xh, g1, b1, g2, b2, Wq, Wk, Wv1, Wv2, out_l, out_h);
}
